// EdgeAttention_54597624267069
// MI455X (gfx1250) — matrix operand order repaired and checked by reading, not compiled, not run on hardware
//
#include <hip/hip_runtime.h>
#include <hip/hip_bf16.h>

#define N_EDGES   800000
#define NODE_DIM  128
#define EDGE_DIM  64
#define HIDDEN    64
#define IN_DIM    320
#define LN_EPS    1e-5f

typedef __attribute__((ext_vector_type(16))) __bf16       v16bf;
typedef __attribute__((ext_vector_type(8)))  float        v8f;
typedef __attribute__((ext_vector_type(4)))  unsigned int v4u;
typedef __attribute__((ext_vector_type(8)))  int          v8i;
typedef __attribute__((ext_vector_type(4)))  int          v4i;

// ---------------------------------------------------------------------------
// Pre-pass: repack W1 (f32 [320][64], row-major K x N) into bf16 fragments in
// a contiguous-k WMMA B-operand order (k = 16h + j, same as A), fragment-major:
//   fragment f = ks*4 + nt  (ks = K-step 0..9, nt = N-tile 0..3), 512 bf16 each
//   lane L (0..31), element j (0..15):
//     ncol = nt*16 + (L & 15)
//     k    = ks*32 + (L >= 16 ? 16 : 0) + j
// so each lane's 32 bytes are contiguous at f*1024 + L*32.
// ---------------------------------------------------------------------------
__global__ void w1_pack_kernel(const float* __restrict__ W1,
                               __bf16* __restrict__ wpack) {
    int tid = blockIdx.x * blockDim.x + threadIdx.x;
    if (tid >= 40 * 512) return;
    int f  = tid >> 9;          // fragment 0..39
    int L  = (tid >> 4) & 31;   // lane 0..31
    int j  = tid & 15;          // element 0..15
    int ks = f >> 2;
    int nt = f & 3;
    int ncol = nt * 16 + (L & 15);
    int k    = ks * 32 + ((L >= 16) ? 16 : 0) + j;
    wpack[tid] = (__bf16)W1[k * HIDDEN + ncol];
}

// ---------------------------------------------------------------------------
// Main kernel: 256 threads = 8 waves per block; each wave owns a 16-edge tile.
//  - block: stage packed W1 (40 KB bf16) into LDS via the Tensor Data Mover
//           (wave 0 issues tensor_load_to_lds, waits TENSORcnt, block barrier)
//  - wave:  10 K-steps; per step each lane gathers its A-fragment straight
//           from global (layout-aware), converts to bf16, 4 WMMAs vs LDS B
//  - epilogue: C tile -> per-wave LDS scratch (68-float row stride, bank-
//           conflict free), each lane reduces half a row; 3 xor-16 shuffles.
// ---------------------------------------------------------------------------
__global__ __launch_bounds__(256)
void edge_attention_kernel(const float* __restrict__ x,
                           const int*   __restrict__ edge_index,
                           const float* __restrict__ edge_attr,
                           const __bf16* __restrict__ wpack,
                           const float* __restrict__ b1,
                           const float* __restrict__ gamma,
                           const float* __restrict__ beta,
                           const float* __restrict__ W2,
                           const float* __restrict__ b2,
                           float* __restrict__ out) {
    __shared__ __align__(32) __bf16 lds_w1[40 * 512];     // 40 KB  B fragments
    __shared__ __align__(16) float  lds_h[8][16 * 68 + 4]; // ~34 KB C scratch

    // ---- stage packed W1 into LDS -------------------------------------
#if __has_builtin(__builtin_amdgcn_tensor_load_to_lds)
    if (threadIdx.x < 32) {                 // wave 0 issues the TDM op
        unsigned ldsa = (unsigned)(uintptr_t)(&lds_w1[0]);
        unsigned long long ga = (unsigned long long)(uintptr_t)wpack;
        // D# group0: count=1 (valid user descriptor), lds_addr, global_addr,
        //            type=2 ("image") in bits 127:126
        v4u g0 = { 1u,
                   ldsa,
                   (unsigned)(ga & 0xFFFFFFFFu),
                   (unsigned)((ga >> 32) & 0x1FFFFFFu) | (2u << 30) };
        // D# group1: 1-D tensor of 5120 x 8-byte elements (= 40960 bytes)
        //   data_size=3 (8B) @bits17:16; tensor_dim0=5120 @79:48;
        //   tensor_dim1=1 @111:80; tile_dim0=5120 @127:112;
        //   tile_dim1=0 (unused); tensor_dim0_stride=5120 @207:160
        v8i g1 = { (int)(3u << 16),
                   (int)((5120u & 0xFFFFu) << 16),
                   (int)((5120u >> 16) | (1u << 16)),
                   (int)(5120u << 16),
                   0,
                   (int)5120u,
                   0, 0 };
        v4i gz = { 0, 0, 0, 0 };
#if defined(__clang_major__) && (__clang_major__ >= 23)
        v8i gz8 = { 0, 0, 0, 0, 0, 0, 0, 0 };
        __builtin_amdgcn_tensor_load_to_lds(g0, g1, gz, gz, gz8, 0);
#else
        __builtin_amdgcn_tensor_load_to_lds(g0, g1, gz, gz, 0);
#endif
        __builtin_amdgcn_s_wait_tensorcnt(0);
    }
    __syncthreads();
#else
    {
        const uint4* s = (const uint4*)wpack;
        uint4*       d = (uint4*)lds_w1;
        for (int i = threadIdx.x; i < 2560; i += 256) d[i] = s[i];
    }
    __syncthreads();
#endif

    const int lane = threadIdx.x & 31;
    const int wave = threadIdx.x >> 5;
    const int tile = blockIdx.x * 8 + wave;
    const int ln   = lane & 15;
    const int e    = tile * 16 + ln;               // this lane's edge (row m = ln)

    const int srcn = edge_index[e];
    const int dstn = edge_index[N_EDGES + e];
    const float* rs = x + (size_t)srcn * NODE_DIM;
    const float* rd = x + (size_t)dstn * NODE_DIM;
    const float* re = edge_attr + (size_t)e * EDGE_DIM;
    const int kk = (lane < 16) ? 0 : 16;           // A-layout K sub-offset

    // accumulators pre-loaded with bias (bias depends only on column)
    float b1v[4];
    #pragma unroll
    for (int nt = 0; nt < 4; ++nt) b1v[nt] = b1[nt * 16 + ln];
    const float b2v = b2[0];

    v8f acc[4];
    #pragma unroll
    for (int nt = 0; nt < 4; ++nt)
        #pragma unroll
        for (int r = 0; r < 8; ++r)
            acc[nt][r] = b1v[nt];

    #pragma unroll
    for (int ks = 0; ks < 10; ++ks) {
        const int kbase = ks * 32;
        // each 32-wide K window lies entirely inside one source region
        const float* p = (kbase < 128) ? (rs + kbase)
                       : (kbase < 256) ? (rd + (kbase - 128))
                       :                 (re + (kbase - 256));
        // lane's A data: K = kk..kk+15 (contiguous, matching packed B)
        float4 c0 = *(const float4*)(p + kk);
        float4 c1 = *(const float4*)(p + kk + 4);
        float4 c2 = *(const float4*)(p + kk + 8);
        float4 c3 = *(const float4*)(p + kk + 12);
        v16bf a;
        a[0]  = (__bf16)c0.x; a[1]  = (__bf16)c0.y; a[2]  = (__bf16)c0.z; a[3]  = (__bf16)c0.w;
        a[4]  = (__bf16)c1.x; a[5]  = (__bf16)c1.y; a[6]  = (__bf16)c1.z; a[7]  = (__bf16)c1.w;
        a[8]  = (__bf16)c2.x; a[9]  = (__bf16)c2.y; a[10] = (__bf16)c2.z; a[11] = (__bf16)c2.w;
        a[12] = (__bf16)c3.x; a[13] = (__bf16)c3.y; a[14] = (__bf16)c3.z; a[15] = (__bf16)c3.w;

        #pragma unroll
        for (int nt = 0; nt < 4; ++nt) {
            const v16bf* bp =
                (const v16bf*)(lds_w1 + (ks * 4 + nt) * 512 + lane * 16);
            v16bf bfrag = *bp;
            acc[nt] = __builtin_amdgcn_wmma_f32_16x16x32_bf16(
                /*neg_a=*/false, a, /*neg_b=*/false, bfrag,
                /*c_mod=*/(short)0, acc[nt],
                /*reuse_a=*/false, /*reuse_b=*/false);
        }
    }

    // ---- epilogue: LDS transpose, per-lane half-row LayerNorm ----------
    // C/D layout: lanes 0-15 hold rows r (element r), lanes 16-31 rows r+8;
    // column n = nt*16 + (lane&15). Row stride 68 floats => per-instruction
    // bank-conflict-free stores and b128 loads.
    float* hs = &lds_h[wave][0];
    #pragma unroll
    for (int nt = 0; nt < 4; ++nt)
        #pragma unroll
        for (int r = 0; r < 8; ++r) {
            int row = (lane < 16) ? r : (r + 8);
            hs[row * 68 + nt * 16 + ln] = acc[nt][r];
        }

    const int hh = lane >> 4;                 // half: columns hh*32 .. +31
    const float* hr = hs + ln * 68 + hh * 32; // this lane's half-row
    float hv[32];
    #pragma unroll
    for (int c = 0; c < 8; ++c) {
        float4 t = *(const float4*)(hr + 4 * c);
        hv[4*c+0] = t.x; hv[4*c+1] = t.y; hv[4*c+2] = t.z; hv[4*c+3] = t.w;
    }
    float s = 0.0f, q = 0.0f;
    #pragma unroll
    for (int c = 0; c < 32; ++c) { s += hv[c]; q = fmaf(hv[c], hv[c], q); }
    s += __shfl_xor(s, 16, 32);
    q += __shfl_xor(q, 16, 32);
    float mu  = s * (1.0f / 64.0f);
    float var = q * (1.0f / 64.0f) - mu * mu;
    float inv = rsqrtf(var + LN_EPS);

    const float* gp = gamma + hh * 32;
    const float* bp = beta  + hh * 32;
    const float* wp = W2    + hh * 32;
    float d = 0.0f;
    #pragma unroll
    for (int c = 0; c < 8; ++c) {
        float4 g4 = *(const float4*)(gp + 4 * c);
        float4 e4 = *(const float4*)(bp + 4 * c);
        float4 w4 = *(const float4*)(wp + 4 * c);
        d += fmaxf((hv[4*c+0] - mu) * inv * g4.x + e4.x, 0.0f) * w4.x;
        d += fmaxf((hv[4*c+1] - mu) * inv * g4.y + e4.y, 0.0f) * w4.y;
        d += fmaxf((hv[4*c+2] - mu) * inv * g4.z + e4.z, 0.0f) * w4.z;
        d += fmaxf((hv[4*c+3] - mu) * inv * g4.w + e4.w, 0.0f) * w4.w;
    }
    d += __shfl_xor(d, 16, 32);
    if (lane < 16) {
        float z = d + b2v;
        out[tile * 16 + ln] = 1.0f / (1.0f + __expf(-z));
    }
}

// ---------------------------------------------------------------------------
// inputs (setup_inputs order):
//  0:x f32[50000*128] 1:edge_index i32[2*800000] 2:edge_attr f32[800000*64]
//  3:W1 f32[320*64] 4:b1 f32[64] 5:gamma f32[64] 6:beta f32[64]
//  7:W2 f32[64] 8:b2 f32[1];  out: f32[800000]
// ---------------------------------------------------------------------------
extern "C" void kernel_launch(void* const* d_in, const int* in_sizes, int n_in,
                              void* d_out, int out_size, void* d_ws, size_t ws_size,
                              hipStream_t stream) {
    const float* x          = (const float*)d_in[0];
    const int*   edge_index = (const int*)d_in[1];
    const float* edge_attr  = (const float*)d_in[2];
    const float* W1         = (const float*)d_in[3];
    const float* b1         = (const float*)d_in[4];
    const float* gamma      = (const float*)d_in[5];
    const float* beta       = (const float*)d_in[6];
    const float* W2         = (const float*)d_in[7];
    const float* b2         = (const float*)d_in[8];
    __bf16* wpack = (__bf16*)d_ws;   // 40*512 bf16 = 40 KB of workspace

    w1_pack_kernel<<<80, 256, 0, stream>>>(W1, wpack);

    // 800000 edges / (16 per wave * 8 waves per block) = 6250 blocks exactly
    edge_attention_kernel<<<N_EDGES / (16 * 8), 256, 0, stream>>>(
        x, edge_index, edge_attr, wpack, b1, gamma, beta, W2, b2,
        (float*)d_out);
}
